// Decoder_30940944400612
// MI455X (gfx1250) — compile-verified, matrix operand order checked by reading, not run on hardware
//
#include <hip/hip_runtime.h>
#include <cstdint>
#include <cstddef>

#define B_ 512
#define L_ 256
#define H_ 256
#define G3H_ 768   // 3*H

typedef __bf16 bf16;
typedef __attribute__((ext_vector_type(16))) __bf16 v16bf;
typedef __attribute__((ext_vector_type(8)))  __bf16 bf16x8;
typedef __attribute__((ext_vector_type(8)))  float   v8f;

__device__ __forceinline__ bf16  f2bf(float f) { return (bf16)f; }
__device__ __forceinline__ float bf2f(bf16 b)  { return (float)b; }
__device__ __forceinline__ float sigmoidf(float x) { return 1.f / (1.f + __expf(-x)); }

#define WMMA_BF16(a, b, c) \
  __builtin_amdgcn_wmma_f32_16x16x32_bf16(false, (a), false, (b), (short)0, (c), false, false)

// ---------------------------------------------------------------------------
// Fragment loaders (CDNA5 16-bit 16x32 layout, §7.12.2):
//   lane%16 = row, lane/16 = K-group; elems 0..7 <-> K=kb+kg*8+0..7,
//   elems 8..15 <-> K=kb+16+kg*8+0..7. Two 16-byte chunks per lane.
// ---------------------------------------------------------------------------
__device__ __forceinline__ v16bf load_frag_f32(const float* base, int ld, int r0,
                                               int kb, int lane) {
  int row = r0 + (lane & 15);
  int kg  = (lane >> 4) & 1;
  const float* p = base + (size_t)row * ld + kb + kg * 8;
  float4 f0 = *(const float4*)(p);
  float4 f1 = *(const float4*)(p + 4);
  float4 f2 = *(const float4*)(p + 16);
  float4 f3 = *(const float4*)(p + 20);
  v16bf a;
  a[0]  = f2bf(f0.x); a[1]  = f2bf(f0.y); a[2]  = f2bf(f0.z); a[3]  = f2bf(f0.w);
  a[4]  = f2bf(f1.x); a[5]  = f2bf(f1.y); a[6]  = f2bf(f1.z); a[7]  = f2bf(f1.w);
  a[8]  = f2bf(f2.x); a[9]  = f2bf(f2.y); a[10] = f2bf(f2.z); a[11] = f2bf(f2.w);
  a[12] = f2bf(f3.x); a[13] = f2bf(f3.y); a[14] = f2bf(f3.z); a[15] = f2bf(f3.w);
  return a;
}

// Unguarded bf16 fragment load (statically in-range tiles).
__device__ __forceinline__ v16bf load_frag_bf16(const bf16* base, int ld, int r0,
                                                int kb, int lane) {
  int row = r0 + (lane & 15);
  int kg  = (lane >> 4) & 1;
  const bf16* p = base + (size_t)row * ld + kb + kg * 8;
  bf16x8 lo = *(const bf16x8*)(p);
  bf16x8 hi = *(const bf16x8*)(p + 16);
  v16bf a;
#pragma unroll
  for (int i = 0; i < 8; ++i) { a[i] = lo[i]; a[8 + i] = hi[i]; }
  return a;
}

// Guarded variant (only used for N-tail tiles of the N=40 head GEMM).
__device__ __forceinline__ v16bf load_frag_bf16_g(const bf16* base, int ld, int r0,
                                                  int kb, int lane, int rmax) {
  int row = r0 + (lane & 15);
  int kg  = (lane >> 4) & 1;
  v16bf a;
  if (row < rmax) {
    const bf16* p = base + (size_t)row * ld + kb + kg * 8;
    bf16x8 lo = *(const bf16x8*)(p);
    bf16x8 hi = *(const bf16x8*)(p + 16);
#pragma unroll
    for (int i = 0; i < 8; ++i) { a[i] = lo[i]; a[8 + i] = hi[i]; }
  } else {
#pragma unroll
    for (int i = 0; i < 16; ++i) a[i] = f2bf(0.f);
  }
  return a;
}

// ---------------------------------------------------------------------------
// Generic GEMM: C(MxN,f32) = A(MxK,f32,lda) @ W(NxKp,bf16)^T + bias, epilogue.
// One wave per 16x64 tile. MODE: 0 plain, 1 sigmoid, 2 exp(-relu),
// 3 time-embed (aux0=times, aux1=W_te, aux2=b_te), 4 add aux0[(m/L)*N+n].
// GUARD selected at host: only needed when N % 64 != 0.
// ---------------------------------------------------------------------------
template <int MODE, bool GUARD>
__global__ __launch_bounds__(256) void k_gemm(
    const float* __restrict__ A, int lda, const bf16* __restrict__ W, int Kp,
    const float* __restrict__ bias, float* __restrict__ C, int M, int N,
    int ntq, int L, const float* __restrict__ aux0,
    const float* __restrict__ aux1, const float* __restrict__ aux2) {
  int wv   = blockIdx.x * 8 + (threadIdx.x >> 5);
  int lane = threadIdx.x & 31;
  int total = (M >> 4) * ntq;
  if (wv >= total) return;                 // wave-uniform: EXEC stays all-ones
  int mt = wv / ntq, nq = wv - mt * ntq;
  int m0 = mt << 4, nb = nq << 6;

  v8f c[4];
#pragma unroll
  for (int t = 0; t < 4; ++t)
#pragma unroll
    for (int r = 0; r < 8; ++r) c[t][r] = 0.f;

  for (int kb = 0; kb < Kp; kb += 32) {
    v16bf a = load_frag_f32(A, lda, m0, kb, lane);
    v16bf bt[4];
#pragma unroll
    for (int t = 0; t < 4; ++t)
      bt[t] = GUARD ? load_frag_bf16_g(W, Kp, nb + t * 16, kb, lane, N)
                    : load_frag_bf16(W, Kp, nb + t * 16, kb, lane);
#pragma unroll
    for (int t = 0; t < 4; ++t) c[t] = WMMA_BF16(a, bt[t], c[t]);
  }

  int colb = lane & 15;
  int rowo = (lane >> 4) << 3;
#pragma unroll
  for (int t = 0; t < 4; ++t) {
    int n = nb + t * 16 + colb;
    if (!GUARD || n < N) {
      float bi = bias ? bias[n] : 0.f;
#pragma unroll
      for (int r = 0; r < 8; ++r) {
        int m = m0 + rowo + r;
        float v = c[t][r] + bi;
        if (MODE == 1) {
          v = sigmoidf(v);
        } else if (MODE == 2) {
          v = __expf(-fmaxf(v, 0.f));
        } else if (MODE == 3) {
          float pt = (m % L) ? aux0[m - 1] : 0.f;
          float te = aux1[n] * pt + aux2[n];
          v += (n == 0) ? te : __sinf(te);
        } else if (MODE == 4) {
          v += aux0[(size_t)(m / L) * N + n];
        }
        C[(size_t)m * N + n] = v;
      }
    }
  }
}

// ---------------------------------------------------------------------------
// GRU scan. One block = 16 batch rows, 8 waves (blockDim == H == 256).
// Per step: hp(16x768) = h(16x256,bf16 LDS) @ Whh^T (bf16, L2-hot),
// 48 WMMA/wave, then gate math with thread tid owning column j=tid for all
// 16 rows (biases held in registers, seq_len in LDS).
// ---------------------------------------------------------------------------
__global__ __launch_bounds__(256) void k_gru_scan(
    const float* __restrict__ xpre,      // (B,L,3H) f32
    const float* __restrict__ h0, int h0stride,
    const bf16* __restrict__ Whh,        // (3H, H) bf16
    const float* __restrict__ bhh,       // (3H)
    const int* __restrict__ seqlen,      // (B)
    float* __restrict__ out, int L) {    // (B,L,H)
  __shared__ bf16  hsh[16][H_];
  __shared__ float hp[16][G3H_];
  __shared__ int   slen[16];
  int b0   = blockIdx.x << 4;
  int tid  = threadIdx.x;
  int wave = tid >> 5, lane = tid & 31;

  for (int i = tid; i < 16 * H_; i += 256)
    hsh[i >> 8][i & 255] = f2bf(h0[(size_t)(b0 + (i >> 8)) * h0stride + (i & 255)]);
  if (tid < 16) slen[tid] = seqlen[b0 + tid];
  // thread tid handles column j = tid in the gate phase
  float br = bhh[tid], bz = bhh[tid + H_], bn = bhh[tid + 2 * H_];
  __syncthreads();

  for (int l = 0; l < L; ++l) {
    v8f c[6];
#pragma unroll
    for (int t = 0; t < 6; ++t)
#pragma unroll
      for (int r = 0; r < 8; ++r) c[t][r] = 0.f;

#pragma unroll
    for (int kb = 0; kb < H_; kb += 32) {
      v16bf bt[6];
#pragma unroll
      for (int t = 0; t < 6; ++t)
        bt[t] = load_frag_bf16(Whh, H_, (wave * 6 + t) << 4, kb, lane);
      v16bf a = load_frag_bf16(&hsh[0][0], H_, 0, kb, lane);
#pragma unroll
      for (int t = 0; t < 6; ++t) c[t] = WMMA_BF16(a, bt[t], c[t]);
    }

    int colb = lane & 15;
    int rowo = (lane >> 4) << 3;
#pragma unroll
    for (int t = 0; t < 6; ++t) {
      int n = ((wave * 6 + t) << 4) + colb;
#pragma unroll
      for (int r = 0; r < 8; ++r) hp[rowo + r][n] = c[t][r];
    }
    __syncthreads();

#pragma unroll 4
    for (int m = 0; m < 16; ++m) {
      size_t xi = ((size_t)(b0 + m) * L + l) * G3H_ + tid;
      float xr = xpre[xi], xz = xpre[xi + H_], xn = xpre[xi + 2 * H_];
      float hr = hp[m][tid] + br;
      float hz = hp[m][tid + H_] + bz;
      float hn = hp[m][tid + 2 * H_] + bn;
      float r  = sigmoidf(xr + hr);
      float z  = sigmoidf(xz + hz);
      float nc = tanhf(xn + r * hn);
      float hold = bf2f(hsh[m][tid]);
      float hnew = (1.f - z) * nc + z * hold;
      bool  v = l < slen[m];
      hsh[m][tid] = f2bf(v ? hnew : hold);
      out[((size_t)(b0 + m) * L + l) * H_ + tid] = v ? hnew : 0.f;
    }
    __syncthreads();
  }
}

// --------------------------- small helper kernels ---------------------------
__global__ void k_cvt(const float* __restrict__ src, int sld, int soff,
                      bf16* __restrict__ dst, int N, int K, int Kp) {
  int i = blockIdx.x * 256 + threadIdx.x;
  if (i >= N * Kp) return;
  int n = i / Kp, k = i - n * Kp;
  float v = (k < K) ? src[(size_t)n * sld + soff + k] : 0.f;
  dst[i] = f2bf(v);
}

// XCAT (B,L,384): [sta | shifted dynamics | shifted priv | shifted mask | 0pad]
__global__ void k_xcat(const float* __restrict__ sta, const float* __restrict__ dyn,
                       const float* __restrict__ priv, const float* __restrict__ mask,
                       float* __restrict__ x, int L) {
  size_t i = (size_t)blockIdx.x * 256 + threadIdx.x;
  size_t total = (size_t)B_ * L * 384;
  if (i >= total) return;
  int col = (int)(i % 384);
  size_t row = i / 384;          // b*L + l
  int l = (int)(row % L);
  float v = 0.f;
  if (col < 64) {
    v = sta[(row / L) * 64 + col];
  } else if (l > 0) {
    size_t pr = row - 1;
    if      (col < 192) v = dyn [pr * 128 + (col - 64)];
    else if (col < 320) v = priv[pr * 128 + (col - 192)];
    else if (col < 360) v = mask[pr * 40  + (col - 320)];
  }
  x[i] = v;
}

// MLCAT (B,L,96): [mask | lag | 0pad]   (unshifted)
__global__ void k_ml(const float* __restrict__ mask, const float* __restrict__ lag,
                     float* __restrict__ ml) {
  size_t i = (size_t)blockIdx.x * 256 + threadIdx.x;
  size_t total = (size_t)B_ * L_ * 96;
  if (i >= total) return;
  int col = (int)(i % 96);
  size_t row = i / 96;
  float v = 0.f;
  if (col < 40)      v = mask[row * 40 + col];
  else if (col < 80) v = lag [row * 40 + (col - 40)];
  ml[i] = v;
}

__global__ void k_mul(float* __restrict__ a, const float* __restrict__ b, size_t n) {
  size_t i = (size_t)blockIdx.x * 256 + threadIdx.x;
  if (i < n) a[i] *= b[i];
}

// gen_times: wave-per-row dot(out_row, W_tfc) -> sigmoid + pt
__global__ __launch_bounds__(256) void k_times(
    const float* __restrict__ out, const float* __restrict__ wt,
    const float* __restrict__ bt, const float* __restrict__ times,
    float* __restrict__ gt, int L) {
  int row  = blockIdx.x * 8 + (threadIdx.x >> 5);
  int lane = threadIdx.x & 31;
  if (row >= B_ * L) return;
  const float* o = out + (size_t)row * H_;
  float s = 0.f;
  for (int j = lane; j < H_; j += 32) s += o[j] * wt[j];
#pragma unroll
  for (int off = 16; off > 0; off >>= 1) s += __shfl_down(s, off, 32);
  if (lane == 0) {
    float pt = (row % L) ? times[row - 1] : 0.f;
    gt[row] = sigmoidf(s + bt[0]) + pt;
  }
}

// act_dynamics in-place on (rows,128): 8 groups of 12 (softmax 11 + sigmoid 1),
// then sigmoid on cols 96..127.
__global__ void k_act_dyn(float* __restrict__ y, int rows) {
  int r = blockIdx.x * 256 + threadIdx.x;
  if (r >= rows) return;
  float* p = y + (size_t)r * 128;
  for (int g = 0; g < 8; ++g) {
    float* q = p + g * 12;
    float mx = q[0];
    for (int j = 1; j < 11; ++j) mx = fmaxf(mx, q[j]);
    float s = 0.f;
    for (int j = 0; j < 11; ++j) { float e = __expf(q[j] - mx); q[j] = e; s += e; }
    float inv = 1.f / s;
    for (int j = 0; j < 11; ++j) q[j] *= inv;
    q[11] = sigmoidf(q[11]);
  }
  for (int j = 96; j < 128; ++j) p[j] = sigmoidf(p[j]);
}

// act_statics in-place on (B,64): 4 groups of 8 softmax, sigmoid 32..63.
__global__ void k_act_sta(float* __restrict__ y, int rows) {
  int r = blockIdx.x * 256 + threadIdx.x;
  if (r >= rows) return;
  float* p = y + (size_t)r * 64;
  for (int g = 0; g < 4; ++g) {
    float* q = p + g * 8;
    float mx = q[0];
    for (int j = 1; j < 8; ++j) mx = fmaxf(mx, q[j]);
    float s = 0.f;
    for (int j = 0; j < 8; ++j) { float e = __expf(q[j] - mx); q[j] = e; s += e; }
    float inv = 1.f / s;
    for (int j = 0; j < 8; ++j) q[j] *= inv;
  }
  for (int j = 32; j < 64; ++j) p[j] = sigmoidf(p[j]);
}

// ---------------------------------------------------------------------------
extern "C" void kernel_launch(void* const* d_in, const int* in_sizes, int n_in,
                              void* d_out, int out_size, void* d_ws, size_t ws_size,
                              hipStream_t stream) {
  (void)in_sizes; (void)n_in; (void)out_size; (void)ws_size;

  const float* embed0   = (const float*)d_in[0];
  const float* mu       = (const float*)d_in[1];
  const float* logvar   = (const float*)d_in[2];
  const float* sta      = (const float*)d_in[3];
  const float* dynamics = (const float*)d_in[4];
  const float* lag      = (const float*)d_in[5];
  const float* mask     = (const float*)d_in[6];
  const float* priv     = (const float*)d_in[7];
  const float* times    = (const float*)d_in[8];
  const int*   seqlen   = (const int*)  d_in[9];
  const float* W_emb  = (const float*)d_in[10]; const float* b_emb  = (const float*)d_in[11];
  const float* W_te   = (const float*)d_in[12]; const float* b_te   = (const float*)d_in[13];
  const float* Wih0   = (const float*)d_in[14]; const float* Whh0   = (const float*)d_in[15];
  const float* bih0   = (const float*)d_in[16]; const float* bhh0   = (const float*)d_in[17];
  const float* Wih1   = (const float*)d_in[18]; const float* Whh1   = (const float*)d_in[19];
  const float* bih1   = (const float*)d_in[20]; const float* bhh1   = (const float*)d_in[21];
  const float* Wih_r  = (const float*)d_in[22]; const float* Whh_r  = (const float*)d_in[23];
  const float* bih_r  = (const float*)d_in[24]; const float* bhh_r  = (const float*)d_in[25];
  const float* W_sta  = (const float*)d_in[26]; const float* b_sta  = (const float*)d_in[27];
  const float* W_dyn  = (const float*)d_in[28]; const float* b_dyn  = (const float*)d_in[29];
  const float* W_dec  = (const float*)d_in[30]; const float* b_dec  = (const float*)d_in[31];
  const float* W_miss = (const float*)d_in[32]; const float* b_miss = (const float*)d_in[33];
  const float* W_tfc  = (const float*)d_in[34]; const float* b_tfc  = (const float*)d_in[35];

  const int M = B_ * L_;   // 131072

  // ---- workspace layout ----
  float* bufP  = (float*)d_ws;                        // 131072*768 (also XCAT 131072*384)
  float* bufA  = bufP  + (size_t)M * G3H_;            // 131072*256
  float* bufB  = bufA  + (size_t)M * H_;              // 131072*256
  float* bufML = bufB  + (size_t)M * H_;              // 131072*96
  float* g3    = bufML + (size_t)M * 96;              // 512*768
  bf16* wb = (bf16*)(g3 + (size_t)B_ * G3H_);
  bf16* Wemb_b = wb;  wb += 256 * 384;
  bf16* Wih0_b = wb;  wb += 768 * 256;
  bf16* Whh0_b = wb;  wb += 768 * 256;
  bf16* Wih1_b = wb;  wb += 768 * 256;
  bf16* Whh1_b = wb;  wb += 768 * 256;
  bf16* Wr1_b  = wb;  wb += 768 * 256;
  bf16* Wr2_b  = wb;  wb += 768 * 256;
  bf16* Whhr_b = wb;  wb += 768 * 256;
  bf16* Wsta_b = wb;  wb += 64 * 256;
  bf16* Wdyn_b = wb;  wb += 128 * 256;
  bf16* Wdec_b = wb;  wb += 256 * 96;
  bf16* Wmiss_b = wb; wb += 40 * 256;

  // ---- output layout (flat, return order) ----
  float* out_sta  = (float*)d_out;                     // 512*64
  float* out_dyn  = out_sta  + (size_t)B_ * 64;        // 131072*128
  float* out_mask = out_dyn  + (size_t)M * 128;        // 131072*40
  float* out_t    = out_mask + (size_t)M * 40;         // 131072
  float* out_mu   = out_t    + (size_t)M;              // 131072
  float* out_lv   = out_mu   + (size_t)B_ * H_;        // 131072

  auto cvt = [&](const float* s, int sld, int soff, bf16* dst, int N, int K, int Kp) {
    int total = N * Kp;
    k_cvt<<<(total + 255) / 256, 256, 0, stream>>>(s, sld, soff, dst, N, K, Kp);
  };
  cvt(W_emb,  360, 0,   Wemb_b, 256, 360, 384);
  cvt(Wih0,   256, 0,   Wih0_b, 768, 256, 256);
  cvt(Whh0,   256, 0,   Whh0_b, 768, 256, 256);
  cvt(Wih1,   256, 0,   Wih1_b, 768, 256, 256);
  cvt(Whh1,   256, 0,   Whh1_b, 768, 256, 256);
  cvt(Wih_r,  512, 0,   Wr1_b,  768, 256, 256);
  cvt(Wih_r,  512, 256, Wr2_b,  768, 256, 256);
  cvt(Whh_r,  256, 0,   Whhr_b, 768, 256, 256);
  cvt(W_sta,  256, 0,   Wsta_b, 64,  256, 256);
  cvt(W_dyn,  256, 0,   Wdyn_b, 128, 256, 256);
  cvt(W_dec,  80,  0,   Wdec_b, 256, 80,  96);
  cvt(W_miss, 256, 0,   Wmiss_b, 40, 256, 256);

  auto gemm = [&](int mode, const float* A, int lda, const bf16* W, int Kp,
                  const float* bias, float* C, int Mm, int Nn,
                  const float* a0, const float* a1, const float* a2) {
    int ntq = (Nn + 63) / 64;
    long long waves = (long long)(Mm / 16) * ntq;
    dim3 g((unsigned)((waves + 7) / 8)), blk(256);
    bool guard = (Nn % 64) != 0;
    if (!guard) {
      switch (mode) {
        case 0: k_gemm<0,false><<<g, blk, 0, stream>>>(A, lda, W, Kp, bias, C, Mm, Nn, ntq, L_, a0, a1, a2); break;
        case 1: k_gemm<1,false><<<g, blk, 0, stream>>>(A, lda, W, Kp, bias, C, Mm, Nn, ntq, L_, a0, a1, a2); break;
        case 2: k_gemm<2,false><<<g, blk, 0, stream>>>(A, lda, W, Kp, bias, C, Mm, Nn, ntq, L_, a0, a1, a2); break;
        case 3: k_gemm<3,false><<<g, blk, 0, stream>>>(A, lda, W, Kp, bias, C, Mm, Nn, ntq, L_, a0, a1, a2); break;
        case 4: k_gemm<4,false><<<g, blk, 0, stream>>>(A, lda, W, Kp, bias, C, Mm, Nn, ntq, L_, a0, a1, a2); break;
      }
    } else {
      switch (mode) {
        case 0: k_gemm<0,true><<<g, blk, 0, stream>>>(A, lda, W, Kp, bias, C, Mm, Nn, ntq, L_, a0, a1, a2); break;
        case 1: k_gemm<1,true><<<g, blk, 0, stream>>>(A, lda, W, Kp, bias, C, Mm, Nn, ntq, L_, a0, a1, a2); break;
        case 2: k_gemm<2,true><<<g, blk, 0, stream>>>(A, lda, W, Kp, bias, C, Mm, Nn, ntq, L_, a0, a1, a2); break;
        case 3: k_gemm<3,true><<<g, blk, 0, stream>>>(A, lda, W, Kp, bias, C, Mm, Nn, ntq, L_, a0, a1, a2); break;
        case 4: k_gemm<4,true><<<g, blk, 0, stream>>>(A, lda, W, Kp, bias, C, Mm, Nn, ntq, L_, a0, a1, a2); break;
      }
    }
  };
  auto scan = [&](const float* xpre, const float* h0, const bf16* Whh,
                  const float* bhh, float* out) {
    k_gru_scan<<<B_ / 16, 256, 0, stream>>>(xpre, h0, (1 + 3) * H_, Whh, bhh,
                                            seqlen, out, L_);
  };

  // 1) XCAT -> bufP ; X = XCAT @ W_emb^T + b_emb + te -> bufA
  { size_t tot = (size_t)M * 384;
    k_xcat<<<(unsigned)((tot + 255) / 256), 256, 0, stream>>>(sta, dynamics, priv, mask, bufP, L_); }
  gemm(3, bufP, 384, Wemb_b, 384, b_emb, bufA, M, 256, times, W_te, b_te);

  // 2) GRU layer 0
  gemm(0, bufA, 256, Wih0_b, 256, bih0, bufP, M, 768, nullptr, nullptr, nullptr);
  scan(bufP, embed0 + H_, Whh0_b, bhh0, bufA);            // out0 -> bufA

  // 3) GRU layer 1
  gemm(0, bufA, 256, Wih1_b, 256, bih1, bufP, M, 768, nullptr, nullptr, nullptr);
  scan(bufP, embed0 + 2 * H_, Whh1_b, bhh1, bufB);        // out -> bufB

  // 4) heads off layer-1 output
  k_times<<<M / 8, 256, 0, stream>>>(bufB, W_tfc, b_tfc, times, out_t, L_);
  gemm(1, bufB, 256, Wmiss_b, 256, b_miss, out_mask, M, 40, nullptr, nullptr, nullptr);

  // 5) beta = exp(-relu([mask|lag] @ W_dec^T + b_dec)) ; bo = beta * out
  { size_t tot = (size_t)M * 96;
    k_ml<<<(unsigned)((tot + 255) / 256), 256, 0, stream>>>(mask, lag, bufML); }
  gemm(2, bufML, 96, Wdec_b, 96, b_dec, bufA, M, 256, nullptr, nullptr, nullptr);
  { size_t tot = (size_t)M * H_;
    k_mul<<<(unsigned)((tot + 255) / 256), 256, 0, stream>>>(bufA, bufB, tot); }

  // 6) recon GRU: xpre = bo @ Wr1^T + bih_r + glob @ Wr2^T
  gemm(0, embed0, (1 + 3) * H_, Wr2_b, 256, nullptr, g3, B_, 768, nullptr, nullptr, nullptr);
  gemm(4, bufA, 256, Wr1_b, 256, bih_r, bufP, M, 768, g3, nullptr, nullptr);
  scan(bufP, embed0 + 3 * H_, Whhr_b, bhh_r, bufA);       // out1 -> bufA

  // 7) dyn head
  gemm(0, bufA, 256, Wdyn_b, 256, b_dyn, out_dyn, M, 128, nullptr, nullptr, nullptr);
  k_act_dyn<<<(M + 255) / 256, 256, 0, stream>>>(out_dyn, M);

  // 8) statics head
  gemm(0, embed0, (1 + 3) * H_, Wsta_b, 256, b_sta, out_sta, B_, 64, nullptr, nullptr, nullptr);
  k_act_sta<<<(B_ + 255) / 256, 256, 0, stream>>>(out_sta, B_);

  // 9) mu / logvar passthrough
  (void)hipMemcpyAsync(out_mu, mu,     (size_t)B_ * H_ * sizeof(float), hipMemcpyDeviceToDevice, stream);
  (void)hipMemcpyAsync(out_lv, logvar, (size_t)B_ * H_ * sizeof(float), hipMemcpyDeviceToDevice, stream);
}
